// MultiScaleMINDCosine_30056181137855
// MI455X (gfx1250) — compile-verified, matrix operand order checked by reading, not run on hardware
//
#include <hip/hip_runtime.h>
#include <hip/hip_bf16.h>

typedef __attribute__((ext_vector_type(16))) _Float16 v16h;
typedef __attribute__((ext_vector_type(8)))  _Float16 v8h;
typedef __attribute__((ext_vector_type(8)))  float    v8f;

#define DIMSZ 90
#define VOXELS 729000   // 90^3
#define PLANE 8100      // 90^2

__device__ __forceinline__ int clampi(int v, int lo, int hi) {
    return v < lo ? lo : (v > hi ? hi : v);
}
__device__ __forceinline__ int imin(int a, int b) { return a < b ? a : b; }
__device__ __forceinline__ int imax(int a, int b) { return a > b ? a : b; }
__device__ __forceinline__ int iabs(int a) { return a < 0 ? -a : a; }

// ---------------------------------------------------------------------------
// softmax of the 3 scale weights -> miscf[0..2]
// ---------------------------------------------------------------------------
__global__ void init_misc(const float* __restrict__ sw, float* __restrict__ miscf) {
    float m = fmaxf(fmaxf(sw[0], sw[1]), sw[2]);
    float e0 = expf(sw[0] - m), e1 = expf(sw[1] - m), e2 = expf(sw[2] - m);
    float s = e0 + e1 + e2;
    miscf[0] = e0 / s; miscf[1] = e1 / s; miscf[2] = e2 / s;
}

// ---------------------------------------------------------------------------
// MIND step 1: 12 dilated shifted squared differences (edge-clamped)
// layout out: [b][c][z][y][x]
// ---------------------------------------------------------------------------
__global__ void mind_diff2(const float* __restrict__ img, float* __restrict__ d2,
                           int B, int V) {
    int t = blockIdx.x * 256 + threadIdx.x;
    if (t >= B * V) return;
    int b = t / V; int v = t - b * V;
    int x = v % DIMSZ; int y = (v / DIMSZ) % DIMSZ; int z = v / PLANE;
    const float* ib = img + (size_t)b * V;
    auto samp = [&](int zz, int yy, int xx) {
        zz = clampi(zz, 0, DIMSZ - 1); yy = clampi(yy, 0, DIMSZ - 1); xx = clampi(xx, 0, DIMSZ - 1);
        return ib[(zz * DIMSZ + yy) * DIMSZ + xx];
    };
    float s[6];
    s[0] = samp(z - 2, y, x);      // six[0]=(0,1,1)
    s[1] = samp(z, y, x - 2);      // six[1]=(1,1,0)
    s[2] = samp(z, y - 2, x);      // six[2]=(1,0,1)
    s[3] = samp(z, y, x + 2);      // six[3]=(1,1,2)
    s[4] = samp(z + 2, y, x);      // six[4]=(2,1,1)
    s[5] = samp(z, y + 2, x);      // six[5]=(1,2,1)
    const int i1[12] = {1, 2, 2, 3, 3, 4, 4, 4, 5, 5, 5, 5};
    const int i2[12] = {0, 0, 1, 0, 2, 1, 2, 3, 0, 1, 3, 4};
#pragma unroll
    for (int c = 0; c < 12; ++c) {
        float d = s[i1[c]] - s[i2[c]];
        d2[((size_t)(b * 12 + c)) * V + v] = d * d;
    }
}

// ---------------------------------------------------------------------------
// Separable 5-wide box blur along one axis (edge-clamped), optional scale
// ---------------------------------------------------------------------------
__global__ void blur5(const float* __restrict__ in, float* __restrict__ out,
                      int axis, float scale, int total) {
    int t = blockIdx.x * 256 + threadIdx.x;
    if (t >= total) return;
    int v = t % VOXELS; int pl = t / VOXELS;
    int x = v % DIMSZ; int y = (v / DIMSZ) % DIMSZ; int z = v / PLANE;
    const float* p = in + (size_t)pl * VOXELS;
    float s = 0.f;
#pragma unroll
    for (int o = -2; o <= 2; ++o) {
        int zz = z, yy = y, xx = x;
        if (axis == 0) zz = clampi(z + o, 0, DIMSZ - 1);
        else if (axis == 1) yy = clampi(y + o, 0, DIMSZ - 1);
        else xx = clampi(x + o, 0, DIMSZ - 1);
        s += p[(zz * DIMSZ + yy) * DIMSZ + xx];
    }
    out[t] = s * scale;
}

// ---------------------------------------------------------------------------
// per-voxel: mv = mean_c(ssd) - min_c(ssd); block partial sums of mv
// ---------------------------------------------------------------------------
__global__ void mind_mv(const float* __restrict__ ssd, float* __restrict__ mvbuf,
                        float* __restrict__ partials, int B, int V) {
    int t = blockIdx.x * 256 + threadIdx.x;
    float mv = 0.f;
    if (t < B * V) {
        int b = t / V; int v = t - b * V;
        const float* p = ssd + (size_t)b * 12 * V + v;
        float mn = p[0]; float sm = p[0];
#pragma unroll
        for (int c = 1; c < 12; ++c) {
            float u = p[(size_t)c * V];
            mn = fminf(mn, u); sm += u;
        }
        mv = sm * (1.f / 12.f) - mn;
        mvbuf[t] = mv;
    }
    __shared__ float sh[256];
    sh[threadIdx.x] = mv; __syncthreads();
    for (int st = 128; st > 0; st >>= 1) {
        if (threadIdx.x < st) sh[threadIdx.x] += sh[threadIdx.x + st];
        __syncthreads();
    }
    if (threadIdx.x == 0) partials[blockIdx.x] = sh[0];
}

// deterministic reduce of partials -> mm
__global__ void reduce_mm(const float* __restrict__ partials, int n,
                          float* __restrict__ mmslot, float invCount) {
    __shared__ float sh[256];
    float s = 0.f;
    for (int i = threadIdx.x; i < n; i += 256) s += partials[i];
    sh[threadIdx.x] = s; __syncthreads();
    for (int st = 128; st > 0; st >>= 1) {
        if (threadIdx.x < st) sh[threadIdx.x] += sh[threadIdx.x + st];
        __syncthreads();
    }
    if (threadIdx.x == 0) mmslot[0] = sh[0] * invCount;
}

// ---------------------------------------------------------------------------
// mind = exp(-(ssd - min)/clip(mv)), with channel reorder
// ---------------------------------------------------------------------------
__global__ void mind_exp(const float* __restrict__ ssd, const float* __restrict__ mvbuf,
                         const float* __restrict__ mmslot, float* __restrict__ outp,
                         int B, int V) {
    int t = blockIdx.x * 256 + threadIdx.x;
    if (t >= B * V) return;
    int b = t / V; int v = t - b * V;
    const float* p = ssd + (size_t)b * 12 * V + v;
    float vals[12]; float mn = 3.4e38f;
#pragma unroll
    for (int c = 0; c < 12; ++c) { vals[c] = p[(size_t)c * V]; mn = fminf(mn, vals[c]); }
    float mm = mmslot[0];
    float mvc = fminf(fmaxf(mvbuf[t], mm * 0.001f), mm * 1000.f);
    const int order[12] = {6, 8, 1, 11, 2, 10, 0, 7, 9, 4, 5, 3};
#pragma unroll
    for (int j = 0; j < 12; ++j)
        outp[((size_t)(b * 12 + j)) * V + v] = expf(-(vals[order[j]] - mn) / mvc);
}

// ---------------------------------------------------------------------------
// unfold + L2-normalize + fp16 pack: out[b*N+n][Cpad], pad region zeroed
// k = ((c*p + lz)*p + ly)*p + lx
// ---------------------------------------------------------------------------
__global__ void __launch_bounds__(256)
pack_norm(const float* __restrict__ mind, _Float16* __restrict__ outp,
          int p, int Ng, int C, int Cpad, int V) {
    int Npatch = Ng * Ng * Ng;
    int n = blockIdx.x; int b = n / Npatch; n -= b * Npatch;
    int gx = n % Ng; int gy = (n / Ng) % Ng; int gz = n / (Ng * Ng);
    int p2 = p * p, p3 = p2 * p;
    const float* mb = mind + (size_t)b * 12 * V;
    auto getval = [&](int k) {
        int c = k / p3; int r = k - c * p3;
        int lz = r / p2; int rr = r - lz * p2;
        int ly = rr / p; int lx = rr - ly * p;
        return mb[(size_t)c * V + ((gz * p + lz) * DIMSZ + (gy * p + ly)) * DIMSZ + (gx * p + lx)];
    };
    float ss = 0.f;
    for (int k = threadIdx.x; k < C; k += 256) { float u = getval(k); ss += u * u; }
    __shared__ float sh[256];
    __shared__ float rnorm;
    sh[threadIdx.x] = ss; __syncthreads();
    for (int st = 128; st > 0; st >>= 1) {
        if (threadIdx.x < st) sh[threadIdx.x] += sh[threadIdx.x + st];
        __syncthreads();
    }
    if (threadIdx.x == 0) rnorm = 1.f / fmaxf(sqrtf(sh[0]), 1e-12f);
    __syncthreads();
    _Float16* orow = outp + (size_t)blockIdx.x * Cpad;
    for (int k = threadIdx.x; k < Cpad; k += 256) {
        float u = (k < C) ? getval(k) * rnorm : 0.f;
        orow[k] = (_Float16)u;
    }
}

// ---------------------------------------------------------------------------
// Neighborhood-masked similarity + logsumexp loss, one wave per 16-row tile.
// WMMA f32 = f16 x f16 (16x16x32). A-lane layout per CDNA5 ISA 7.12.2:
// lane L(<16): K {kc+0..7, kc+16..23}; lane L+16: K {kc+8..15, kc+24..31}.
// ---------------------------------------------------------------------------
__global__ void __launch_bounds__(32)
nbr_loss(const _Float16* __restrict__ pf, const _Float16* __restrict__ pm,
         int Ng, int R, int Cpad, int B,
         const float* __restrict__ wbuf, int wi,
         float* __restrict__ slots, int slotBase) {
    const int N = Ng * Ng * Ng;
    const int TX = (Ng + 15) >> 4;
    int t = blockIdx.x;
    int xt = t % TX; t /= TX;
    int yf = t % Ng; t /= Ng;
    int zf = t % Ng; int b = t / Ng;
    int xf0 = xt << 4;

    int lane = threadIdx.x;
    int half = lane >> 4;      // 0: rows/K-low group, 1: rows+8/K-high group
    int col  = lane & 15;

    // A fragment rows: lane's fixed patch = col (clamped for memory safety)
    int nf = (zf * Ng + yf) * Ng + imin(xf0 + col, Ng - 1);
    nf = imin(nf, N - 1);
    const _Float16* aRow = pf + ((size_t)(b * N + nf)) * Cpad + half * 8;

    float rowsum[8], diag[8];
#pragma unroll
    for (int r = 0; r < 8; ++r) { rowsum[r] = 0.f; diag[r] = 0.f; }

    int zlo = imax(zf - R, 0), zhi = imin(zf + R, Ng - 1);
    int ylo = imax(yf - R, 0), yhi = imin(yf + R, Ng - 1);
    int xlo = imax(xf0 - R, 0), xhi = imin(xf0 + 15 + R, Ng - 1);

    for (int zm = zlo; zm <= zhi; ++zm)
    for (int ym = ylo; ym <= yhi; ++ym)
    for (int xm0 = xlo; xm0 <= xhi; xm0 += 16) {
        int xm = xm0 + col;
        int nm = (zm * Ng + ym) * Ng + xm;
        nm = imin(nm, N - 1);
        const _Float16* bRow = pm + ((size_t)(b * N + nm)) * Cpad + half * 8;
        __builtin_prefetch(bRow, 0, 1);

        v8f acc;
#pragma unroll
        for (int i = 0; i < 8; ++i) acc[i] = 0.f;

        for (int kc = 0; kc < Cpad; kc += 32) {
            v8h alo = *(const v8h*)(aRow + kc);
            v8h ahi = *(const v8h*)(aRow + kc + 16);
            v8h blo = *(const v8h*)(bRow + kc);
            v8h bhi = *(const v8h*)(bRow + kc + 16);
            v16h av, bv;
#pragma unroll
            for (int i = 0; i < 8; ++i) {
                av[i] = alo[i]; av[i + 8] = ahi[i];
                bv[i] = blo[i]; bv[i + 8] = bhi[i];
            }
            acc = __builtin_amdgcn_wmma_f32_16x16x32_f16(
                false, av, false, bv, (short)0, acc, false, false);
        }

        bool colOk = (xm <= xhi);
#pragma unroll
        for (int r = 0; r < 8; ++r) {
            int M = r + (half << 3);
            int xf = xf0 + M;
            float sim = acc[r];               // sim / TAU, TAU = 1
            bool ok = colOk && (xf < Ng) && (iabs(xf - xm) <= R);
            float e = expf(sim);
            if (ok) rowsum[r] += e;
            if (ok && zm == zf && ym == yf && xm == xf) diag[r] = sim;
        }
    }

    // reduce across the 16 lanes of each half-group (D layout: N = lane&15)
#pragma unroll
    for (int m = 1; m < 16; m <<= 1) {
#pragma unroll
        for (int r = 0; r < 8; ++r) {
            rowsum[r] += __shfl_xor(rowsum[r], m, 16);
            diag[r]   += __shfl_xor(diag[r],   m, 16);
        }
    }

    float part = 0.f;
    if (col == 0) {
#pragma unroll
        for (int r = 0; r < 8; ++r) {
            int xf = xf0 + r + (half << 3);
            if (xf < Ng) part += logf(rowsum[r]) - diag[r];
        }
    }
    part += __shfl_xor(part, 16, 32);   // combine rows 0..7 and 8..15 halves

    if (lane == 0) {
        float coef = wbuf[wi] / (float)(B * N);
        slots[slotBase + blockIdx.x] = part * coef;
    }
}

// deterministic final sum of all tile slots -> scalar output
__global__ void final_reduce(const float* __restrict__ slots, int n, float* __restrict__ out) {
    __shared__ float sh[256];
    float s = 0.f;
    for (int i = threadIdx.x; i < n; i += 256) s += slots[i];
    sh[threadIdx.x] = s; __syncthreads();
    for (int st = 128; st > 0; st >>= 1) {
        if (threadIdx.x < st) sh[threadIdx.x] += sh[threadIdx.x + st];
        __syncthreads();
    }
    if (threadIdx.x == 0) out[0] = sh[0];
}

// ---------------------------------------------------------------------------
extern "C" void kernel_launch(void* const* d_in, const int* in_sizes, int n_in,
                              void* d_out, int out_size, void* d_ws, size_t ws_size,
                              hipStream_t stream) {
    (void)in_sizes; (void)n_in; (void)out_size; (void)ws_size;
    const int B = 2;
    const int V = VOXELS;
    const float* fixedp  = (const float*)d_in[0];
    const float* movingp = (const float*)d_in[1];
    const float* scale_w = (const float*)d_in[2];
    float* out = (float*)d_out;

    char* base = (char*)d_ws;
    auto al = [](size_t x) { return (x + 255) & ~(size_t)255; };
    size_t mindBytes = (size_t)B * 12 * V * sizeof(float);
    size_t o = 0;
    float* mind_f  = (float*)(base + o); o = al(o + mindBytes);
    float* mind_m  = (float*)(base + o); o = al(o + mindBytes);
    float* scrA    = (float*)(base + o); o = al(o + mindBytes);
    float* scrB    = (float*)(base + o); o = al(o + mindBytes);
    float* mvbuf   = (float*)(base + o); o = al(o + (size_t)B * V * sizeof(float));
    float* partials = (float*)(base + o); o = al(o + 8192 * sizeof(float));
    float* slots    = (float*)(base + o); o = al(o + 8192 * sizeof(float));
    float* miscf    = (float*)(base + o); o = al(o + 256);

    init_misc<<<1, 1, 0, stream>>>(scale_w, miscf);

    const float* imgs[2] = {fixedp, movingp};
    float* minds[2] = {mind_f, mind_m};
    int nvox = B * V;
    int g1 = (nvox + 255) / 256;
    int nch = B * 12 * V;
    int g2 = (nch + 255) / 256;
    for (int im = 0; im < 2; ++im) {
        mind_diff2<<<g1, 256, 0, stream>>>(imgs[im], scrA, B, V);
        blur5<<<g2, 256, 0, stream>>>(scrA, scrB, 0, 1.f, nch);
        blur5<<<g2, 256, 0, stream>>>(scrB, scrA, 1, 1.f, nch);
        blur5<<<g2, 256, 0, stream>>>(scrA, scrB, 2, 1.f / 125.f, nch);
        mind_mv<<<g1, 256, 0, stream>>>(scrB, mvbuf, partials, B, V);
        reduce_mm<<<1, 256, 0, stream>>>(partials, g1, miscf + 3, 1.f / (float)nvox);
        mind_exp<<<g1, 256, 0, stream>>>(scrB, mvbuf, miscf + 3, minds[im], B, V);
    }

    const int ps[3] = {3, 5, 9};
    const int rs[3] = {3, 2, 1};
    int slotBase = 0;
    for (int i = 0; i < 3; ++i) {
        int p = ps[i], R = rs[i];
        int Ng = DIMSZ / p;
        int N = Ng * Ng * Ng;
        int C = 12 * p * p * p;
        int Cpad = ((C + 31) / 32) * 32;
        _Float16* pfb = (_Float16*)scrA;
        _Float16* pmb = (_Float16*)scrB;
        pack_norm<<<B * N, 256, 0, stream>>>(mind_f, pfb, p, Ng, C, Cpad, V);
        pack_norm<<<B * N, 256, 0, stream>>>(mind_m, pmb, p, Ng, C, Cpad, V);
        int TX = (Ng + 15) / 16;
        int ntiles = B * Ng * Ng * TX;
        nbr_loss<<<ntiles, 32, 0, stream>>>(pfb, pmb, Ng, R, Cpad, B,
                                            miscf, i, slots, slotBase);
        slotBase += ntiles;
    }
    final_reduce<<<1, 256, 0, stream>>>(slots, slotBase, out);
}
